// Decoder_4535485464632
// MI455X (gfx1250) — matrix operand order repaired and checked by reading, not compiled, not run on hardware
//
#include <hip/hip_runtime.h>
#include <hip/hip_bf16.h>

// Problem dims (match reference)
#define B_ 256
#define S_ 128
#define H_ 512

typedef __attribute__((ext_vector_type(16))) __bf16 v16bf;
typedef __attribute__((ext_vector_type(8)))  __bf16 v8bf;
typedef __attribute__((ext_vector_type(8)))  float  v8f;

// ---------------------------------------------------------------------------
// WMMA strip mainloop: one wave computes a 16 x (NT*16) strip of D.
//   D = A(row-major, bf16) x B^T(row-major N x K, bf16)
// Fragment layouts: k contiguous per lane half (k = 16h + s) in A and B (see previous rounds).
// K is a compile-time constant and the loop is FULLY UNROLLED: straight-line
// code lets the scheduler rename registers across iterations (no rotation
// moves, no rolled-loop v_mov storm) and hoist load clauses ahead of the
// WMMA groups for software pipelining.
// ---------------------------------------------------------------------------
template <int NT, int K>
__device__ __forceinline__ void wmma_strip(const __bf16* __restrict__ A, int lda,
                                           const __bf16* __restrict__ BT, int ldb,
                                           int row0, int col0, int lane,
                                           v8f acc[NT]) {
  const __bf16* ap = A + (size_t)(row0 + (lane & 15)) * lda + ((lane >> 4) * 16);
  const __bf16* bp[NT];
#pragma unroll
  for (int j = 0; j < NT; ++j)
    bp[j] = BT + (size_t)(col0 + j * 16 + (lane & 15)) * ldb + ((lane >> 4) * 16);
#pragma unroll
  for (int k = 0; k < K; k += 32) {
    v8bf alo = *(const v8bf*)(ap + k);
    v8bf ahi = *(const v8bf*)(ap + k + 8);
    v16bf a = __builtin_shufflevector(alo, ahi, 0,1,2,3,4,5,6,7,8,9,10,11,12,13,14,15);
#pragma unroll
    for (int j = 0; j < NT; ++j) {
      __builtin_prefetch(bp[j] + k + 512, 0, 3);   // weight stream toward WGP
      v16bf b = *(const v16bf*)(bp[j] + k);
      acc[j] = __builtin_amdgcn_wmma_f32_16x16x32_bf16(false, a, false, b,
                                                       (short)0, acc[j], false, false);
    }
  }
}

__device__ __forceinline__ float sigf(float x) { return 1.0f / (1.0f + __expf(-x)); }

// ---------------------------------------------------------------------------
// Prep kernels: bf16 weight transposes + state init
// ---------------------------------------------------------------------------
__global__ void prep_wih(const float* __restrict__ Wi, const float* __restrict__ Wh,
                         __bf16* __restrict__ WihT) {
  int i = blockIdx.x * blockDim.x + threadIdx.x;       // over (4H) x (2H)
  if (i >= 4 * H_ * 2 * H_) return;
  int n = i / (2 * H_), k = i % (2 * H_);
  float w = (k < H_) ? Wi[(size_t)k * (4 * H_) + n] : Wh[(size_t)(k - H_) * (4 * H_) + n];
  WihT[i] = (__bf16)w;
}

__global__ void prep_transpose(const float* __restrict__ W, __bf16* __restrict__ WT,
                               int K, int N) {                  // WT[n*K+k] = W[k*N+n]
  int i = blockIdx.x * blockDim.x + threadIdx.x;
  if (i >= K * N) return;
  int n = i / K, k = i % K;
  WT[i] = (__bf16)W[(size_t)k * N + n];
}

__global__ void prep_f2bf(const float* __restrict__ src, __bf16* __restrict__ dst, int n) {
  for (int i = blockIdx.x * blockDim.x + threadIdx.x; i < n; i += gridDim.x * blockDim.x)
    dst[i] = (__bf16)src[i];
}

__global__ void prep_state(const float* __restrict__ x0, const float* __restrict__ h0,
                           const float* __restrict__ c0, __bf16* __restrict__ xh,
                           float* __restrict__ c, float* __restrict__ mask) {
  int i = blockIdx.x * blockDim.x + threadIdx.x;
  if (i < B_ * H_) {
    int b = i / H_, h = i % H_;
    xh[(size_t)b * (2 * H_) + h]       = (__bf16)x0[i];
    xh[(size_t)b * (2 * H_) + H_ + h]  = (__bf16)h0[i];
    c[i] = c0[i];
  }
  if (i < B_ * S_) mask[i] = 1.0f;
}

// ---------------------------------------------------------------------------
// GEMM kernels: 256 threads = 8 waves, each wave owns a 16x64 strip (4 tiles)
// ---------------------------------------------------------------------------
__global__ __launch_bounds__(256) void ctxproj_gemm(const __bf16* __restrict__ ctxbf,
                                                    const __bf16* __restrict__ WctxT,
                                                    __bf16* __restrict__ ctxp) {
  int lane = threadIdx.x & 31, wave = threadIdx.x >> 5;
  int row0 = blockIdx.x * 16;          // M = B*S = 32768
  int col0 = wave * 64;                // N = H = 512 -> 8 strips
  v8f acc[4] = {};
  wmma_strip<4, H_>(ctxbf, H_, WctxT, H_, row0, col0, lane, acc);
  int rbase = row0 + ((lane >> 4) * 8);
#pragma unroll
  for (int j = 0; j < 4; ++j) {
    int col = col0 + j * 16 + (lane & 15);
#pragma unroll
    for (int r = 0; r < 8; ++r)
      ctxp[(size_t)(rbase + r) * H_ + col] = (__bf16)acc[j][r];
  }
}

__global__ __launch_bounds__(256) void gates_gemm(const __bf16* __restrict__ xh,
                                                  const __bf16* __restrict__ WihT,
                                                  const float* __restrict__ bi,
                                                  const float* __restrict__ bh,
                                                  float* __restrict__ gates) {
  int lane = threadIdx.x & 31, wave = threadIdx.x >> 5;
  int row0 = blockIdx.x * 16;                     // M = 256
  int col0 = (blockIdx.y * 8 + wave) * 64;        // N = 2048 -> 32 strips
  v8f acc[4] = {};
  wmma_strip<4, 2 * H_>(xh, 2 * H_, WihT, 2 * H_, row0, col0, lane, acc);
  int rbase = row0 + ((lane >> 4) * 8);
#pragma unroll
  for (int j = 0; j < 4; ++j) {
    int col = col0 + j * 16 + (lane & 15);
    float bias = bi[col] + bh[col];
#pragma unroll
    for (int r = 0; r < 8; ++r)
      gates[(size_t)(rbase + r) * (4 * H_) + col] = acc[j][r] + bias;
  }
}

__global__ __launch_bounds__(256) void q_gemm(const __bf16* __restrict__ hibf,
                                              const __bf16* __restrict__ WhaT,
                                              float* __restrict__ q) {
  int lane = threadIdx.x & 31, wave = threadIdx.x >> 5;
  int row0 = blockIdx.x * 16;          // M = 256
  int col0 = wave * 64;                // N = 512 -> 8 strips
  v8f acc[4] = {};
  wmma_strip<4, H_>(hibf, H_, WhaT, H_, row0, col0, lane, acc);
  int rbase = row0 + ((lane >> 4) * 8);
#pragma unroll
  for (int j = 0; j < 4; ++j) {
    int col = col0 + j * 16 + (lane & 15);
#pragma unroll
    for (int r = 0; r < 8; ++r)
      q[(size_t)(rbase + r) * H_ + col] = acc[j][r];
  }
}

__global__ __launch_bounds__(256) void out_gemm(const __bf16* __restrict__ ah,
                                                const __bf16* __restrict__ WoutT,
                                                const float* __restrict__ bout,
                                                __bf16* __restrict__ xh) {
  int lane = threadIdx.x & 31, wave = threadIdx.x >> 5;
  int row0 = blockIdx.x * 16;          // M = 256
  int col0 = wave * 64;                // N = 512 -> 8 strips
  v8f acc[4] = {};
  wmma_strip<4, 2 * H_>(ah, 2 * H_, WoutT, 2 * H_, row0, col0, lane, acc);
  int rbase = row0 + ((lane >> 4) * 8);
#pragma unroll
  for (int j = 0; j < 4; ++j) {
    int col = col0 + j * 16 + (lane & 15);
    float bb = bout[col];
#pragma unroll
    for (int r = 0; r < 8; ++r)
      xh[(size_t)(rbase + r) * (2 * H_) + H_ + col] = (__bf16)tanhf(acc[j][r] + bb);
  }
}

// ---------------------------------------------------------------------------
// Elementwise / reduction kernels
// ---------------------------------------------------------------------------
__global__ void lstm_cell(const float* __restrict__ gates, float* __restrict__ c,
                          __bf16* __restrict__ hibf, __bf16* __restrict__ ah) {
  int i = blockIdx.x * blockDim.x + threadIdx.x;   // B*H
  if (i >= B_ * H_) return;
  int b = i / H_, n = i % H_;
  const float* g = gates + (size_t)b * (4 * H_);
  float ig = sigf(g[n]);
  float fg = sigf(g[H_ + n]);
  float gg = sigf(g[2 * H_ + n]);     // reference sigmoids the cell gate too
  float og = sigf(g[3 * H_ + n]);
  float ci = fg * c[i] + ig * gg;
  c[i] = ci;
  float hi = og * tanhf(ci);
  hibf[i] = (__bf16)hi;
  ah[(size_t)b * (2 * H_) + H_ + n] = (__bf16)hi;
}

// one wave per (b,s): score = sum_k tanh(ctx_proj + q) * v, masked
__global__ __launch_bounds__(256) void attn_scores(const __bf16* __restrict__ ctxp,
                                                   const float* __restrict__ q,
                                                   const float* __restrict__ v,
                                                   const float* __restrict__ mask,
                                                   float* __restrict__ scores) {
  int lane = threadIdx.x & 31;
  int gid = blockIdx.x * 8 + (threadIdx.x >> 5);   // (b,s) flat, B*S total
  int b = gid >> 7;
  const __bf16* cp = ctxp + (size_t)gid * H_ + lane * 16;
  const float* qp = q + (size_t)b * H_ + lane * 16;
  const float* vp = v + lane * 16;
  v16bf cv = *(const v16bf*)cp;
  float sum = 0.0f;
#pragma unroll
  for (int e = 0; e < 16; ++e)
    sum += tanhf((float)cv[e] + qp[e]) * vp[e];
#pragma unroll
  for (int off = 16; off; off >>= 1) sum += __shfl_xor(sum, off);   // wave32 reduce
  if (lane == 0) scores[gid] = (mask[gid] == 0.0f) ? -1e9f : sum;
}

// one block per batch row: softmax, alpha out, argmax(alpha*mask), mask update
__global__ __launch_bounds__(128) void softmax_pointer(const float* __restrict__ scores,
                                                       float* __restrict__ mask,
                                                       float* __restrict__ alpha,
                                                       int* __restrict__ selIdx,
                                                       float* __restrict__ out_alpha,
                                                       int* __restrict__ out_ptr, int t) {
  __shared__ float red[S_];
  __shared__ int redi[S_];
  int b = blockIdx.x, s = threadIdx.x;
  float sc = scores[b * S_ + s];
  red[s] = sc; __syncthreads();
  for (int off = S_ / 2; off; off >>= 1) {
    if (s < off) red[s] = fmaxf(red[s], red[s + off]);
    __syncthreads();
  }
  float mx = red[0]; __syncthreads();
  float ex = __expf(sc - mx);
  red[s] = ex; __syncthreads();
  for (int off = S_ / 2; off; off >>= 1) {
    if (s < off) red[s] += red[s + off];
    __syncthreads();
  }
  float al = ex / red[0];
  out_alpha[(size_t)b * S_ * S_ + (size_t)t * S_ + s] = al;
  alpha[b * S_ + s] = al;
  float m = mask[b * S_ + s];
  __syncthreads();
  red[s] = al * m; redi[s] = s; __syncthreads();
  for (int off = S_ / 2; off; off >>= 1) {
    if (s < off && red[s + off] > red[s]) { red[s] = red[s + off]; redi[s] = redi[s + off]; }
    __syncthreads();
  }
  int idx = redi[0];
  if (s == 0) { out_ptr[b * S_ + t] = idx; selIdx[b] = idx; }
  if (s == idx) mask[b * S_ + s] = 0.0f;   // mask *= (1 - one_hot)
}

// one block per batch row: att = alpha @ context ; gather next x = embed[b, idx]
__global__ __launch_bounds__(512) void att_gather(const float* __restrict__ alpha,
                                                  const float* __restrict__ context,
                                                  const float* __restrict__ embed,
                                                  const int* __restrict__ selIdx,
                                                  __bf16* __restrict__ ah,
                                                  __bf16* __restrict__ xh) {
  int b = blockIdx.x, h = threadIdx.x;
  const float* al = alpha + b * S_;
  const float* cb = context + (size_t)b * S_ * H_ + h;
  float acc = 0.0f;
#pragma unroll 4
  for (int s = 0; s < S_; ++s) acc += al[s] * cb[(size_t)s * H_];
  ah[(size_t)b * (2 * H_) + h] = (__bf16)acc;
  int id = selIdx[b];
  xh[(size_t)b * (2 * H_) + h] = (__bf16)embed[((size_t)b * S_ + id) * H_ + h];
}

// ---------------------------------------------------------------------------
extern "C" void kernel_launch(void* const* d_in, const int* in_sizes, int n_in,
                              void* d_out, int out_size, void* d_ws, size_t ws_size,
                              hipStream_t stream) {
  const float* embed = (const float*)d_in[0];
  const float* x0    = (const float*)d_in[1];
  const float* h0    = (const float*)d_in[2];
  const float* c0    = (const float*)d_in[3];
  const float* ctx   = (const float*)d_in[4];
  const float* Wi    = (const float*)d_in[5];
  const float* bi    = (const float*)d_in[6];
  const float* Wh    = (const float*)d_in[7];
  const float* bh    = (const float*)d_in[8];
  const float* Wctx  = (const float*)d_in[9];
  const float* Wha   = (const float*)d_in[10];
  const float* v     = (const float*)d_in[11];
  const float* Wout  = (const float*)d_in[12];
  const float* bout  = (const float*)d_in[13];

  char* ws = (char*)d_ws;
  size_t off = 0;
  auto alloc = [&](size_t bytes) {
    void* p = ws + off;
    off = (off + bytes + 255) & ~(size_t)255;
    return p;
  };
  __bf16* ctxbf  = (__bf16*)alloc((size_t)B_ * S_ * H_ * 2);       // 32 MB
  __bf16* ctxp   = (__bf16*)alloc((size_t)B_ * S_ * H_ * 2);       // 32 MB
  __bf16* WihT   = (__bf16*)alloc((size_t)4 * H_ * 2 * H_ * 2);    // 4 MB
  __bf16* WhaT   = (__bf16*)alloc((size_t)H_ * H_ * 2);
  __bf16* WctxT  = (__bf16*)alloc((size_t)H_ * H_ * 2);
  __bf16* WoutT  = (__bf16*)alloc((size_t)H_ * 2 * H_ * 2);
  __bf16* xh     = (__bf16*)alloc((size_t)B_ * 2 * H_ * 2);        // [x | h]
  __bf16* ah     = (__bf16*)alloc((size_t)B_ * 2 * H_ * 2);        // [att | h_i]
  __bf16* hibf   = (__bf16*)alloc((size_t)B_ * H_ * 2);
  float*  gates  = (float*)alloc((size_t)B_ * 4 * H_ * 4);
  float*  qbuf   = (float*)alloc((size_t)B_ * H_ * 4);
  float*  cbuf   = (float*)alloc((size_t)B_ * H_ * 4);
  float*  mask   = (float*)alloc((size_t)B_ * S_ * 4);
  float*  scores = (float*)alloc((size_t)B_ * S_ * 4);
  float*  alpha  = (float*)alloc((size_t)B_ * S_ * 4);
  int*    selIdx = (int*)alloc((size_t)B_ * sizeof(int));

  float* out_alpha = (float*)d_out;                                // (B,S,S)
  int*   out_ptr   = (int*)((float*)d_out + (size_t)B_ * S_ * S_); // (B,S)

  // One-time prep (re-run every call for determinism)
  prep_wih<<<(4 * H_ * 2 * H_ + 255) / 256, 256, 0, stream>>>(Wi, Wh, WihT);
  prep_transpose<<<(H_ * H_ + 255) / 256, 256, 0, stream>>>(Wha, WhaT, H_, H_);
  prep_transpose<<<(H_ * H_ + 255) / 256, 256, 0, stream>>>(Wctx, WctxT, H_, H_);
  prep_transpose<<<(2 * H_ * H_ + 255) / 256, 256, 0, stream>>>(Wout, WoutT, 2 * H_, H_);
  prep_f2bf<<<4096, 256, 0, stream>>>(ctx, ctxbf, B_ * S_ * H_);
  prep_state<<<(B_ * H_ + 255) / 256, 256, 0, stream>>>(x0, h0, c0, xh, cbuf, mask);
  ctxproj_gemm<<<dim3((B_ * S_) / 16, 1), 256, 0, stream>>>(ctxbf, WctxT, ctxp);

  // Sequential decode loop
  for (int t = 0; t < S_; ++t) {
    gates_gemm<<<dim3(B_ / 16, (4 * H_) / (8 * 64)), 256, 0, stream>>>(xh, WihT, bi, bh, gates);
    lstm_cell<<<(B_ * H_ + 255) / 256, 256, 0, stream>>>(gates, cbuf, hibf, ah);
    q_gemm<<<dim3(B_ / 16, 1), 256, 0, stream>>>(hibf, WhaT, qbuf);
    attn_scores<<<(B_ * S_) / 8, 256, 0, stream>>>(ctxp, qbuf, v, mask, scores);
    softmax_pointer<<<B_, S_, 0, stream>>>(scores, mask, alpha, selIdx, out_alpha, out_ptr, t);
    att_gather<<<B_, H_, 0, stream>>>(alpha, ctx, embed, selIdx, ah, xh);
    out_gemm<<<dim3(B_ / 16, 1), 256, 0, stream>>>(ah, WoutT, bout, xh);
  }
}
